// LightGCN_no_w2v_21474836480430
// MI455X (gfx1250) — matrix operand order repaired and checked by reading, not compiled, not run on hardware
//
#include <hip/hip_runtime.h>
#include <hip/hip_bf16.h>

typedef __attribute__((ext_vector_type(16))) _Float16 v16h;
typedef __attribute__((ext_vector_type(8)))  float    v8f;

#define NUM_USERS 100000
#define NUM_ITEMS 50000
#define N_NODES   (NUM_USERS + NUM_ITEMS)
#define D         64
#define N_EDGES   4800000
#define BATCH     16384
#define N_LAYERS  3

// ---------------------------------------------------------------------------
// 1) X = cat(l2norm(user_w), l2norm(item_w)); ego = acc = X.  One wave per row.
// ---------------------------------------------------------------------------
__global__ void k_normalize_init(const float* __restrict__ uw,
                                 const float* __restrict__ iw,
                                 float* __restrict__ ego,
                                 float* __restrict__ acc) {
    int row  = blockIdx.x * 8 + (threadIdx.x >> 5);
    if (row >= N_NODES) return;
    int lane = threadIdx.x & 31;
    const float* src = (row < NUM_USERS) ? (uw + (size_t)row * D)
                                         : (iw + (size_t)(row - NUM_USERS) * D);
    float2 v = ((const float2*)src)[lane];
    float ss = v.x * v.x + v.y * v.y;
    #pragma unroll
    for (int m = 16; m; m >>= 1) ss += __shfl_xor(ss, m, 32);
    float s = 1.0f / fmaxf(sqrtf(ss), 1e-12f);
    float2 o; o.x = v.x * s; o.y = v.y * s;
    ((float2*)(ego + (size_t)row * D))[lane] = o;
    ((float2*)(acc + (size_t)row * D))[lane] = o;
}

// ---------------------------------------------------------------------------
// 2) SpMM: agg[r] += v * ego[c].  One wave per edge, lane owns a float2 slice.
//    Edge index/val loads forced wave-uniform -> scalar memory path.
//    Native f32 L2 atomics (global_atomic_add_f32).
// ---------------------------------------------------------------------------
__global__ void k_spmm(const float* __restrict__ ego,
                       const int* __restrict__ erow,
                       const int* __restrict__ ecol,
                       const float* __restrict__ eval_,
                       float* __restrict__ agg) {
    int e = blockIdx.x * 8 + (threadIdx.x >> 5);
    if (e >= N_EDGES) return;               // wave-uniform test
    e = __builtin_amdgcn_readfirstlane(e);  // force SGPR -> s_load for edge data
    int lane = threadIdx.x & 31;
    int r = erow[e];
    int c = ecol[e];
    float v = eval_[e];
    float2 x = ((const float2*)(ego + (size_t)c * D))[lane];
    float* dst = agg + (size_t)r * D + lane * 2;
    unsafeAtomicAdd(dst,     v * x.x);
    unsafeAtomicAdd(dst + 1, v * x.y);
}

// ---------------------------------------------------------------------------
// 3) ego = agg + agg*ego; acc += ego  (elementwise)
// ---------------------------------------------------------------------------
__global__ void k_combine(const float* __restrict__ agg,
                          float* __restrict__ ego,
                          float* __restrict__ acc,
                          int n) {
    int i = blockIdx.x * blockDim.x + threadIdx.x;
    if (i >= n) return;
    float a = agg[i];
    float e = ego[i];
    float ne = a + a * e;
    ego[i] = ne;
    acc[i] += ne;
}

// ---------------------------------------------------------------------------
// 4) ui = l2norm(acc / 4).  One wave per row.
// ---------------------------------------------------------------------------
__global__ void k_final_norm(const float* __restrict__ acc,
                             float* __restrict__ ui) {
    int row  = blockIdx.x * 8 + (threadIdx.x >> 5);
    if (row >= N_NODES) return;
    int lane = threadIdx.x & 31;
    float2 v = ((const float2*)(acc + (size_t)row * D))[lane];
    v.x *= 0.25f; v.y *= 0.25f;
    float ss = v.x * v.x + v.y * v.y;
    #pragma unroll
    for (int m = 16; m; m >>= 1) ss += __shfl_xor(ss, m, 32);
    float s = 1.0f / fmaxf(sqrtf(ss), 1e-12f);
    float2 o; o.x = v.x * s; o.y = v.y * s;
    ((float2*)(ui + (size_t)row * D))[lane] = o;
}

// ---------------------------------------------------------------------------
// 5) Gather ue/ie, build cat f16 rows [B x 128], and compute GMF logits.
// ---------------------------------------------------------------------------
__global__ void k_prep(const float* __restrict__ ui,
                       const int* __restrict__ uidx,
                       const int* __restrict__ iidx,
                       const float* __restrict__ aff_w,
                       const float* __restrict__ aff_b,
                       _Float16* __restrict__ cat16,
                       float* __restrict__ logits) {
    int b = blockIdx.x * 8 + (threadIdx.x >> 5);
    if (b >= BATCH) return;
    int lane = threadIdx.x & 31;
    int u = uidx[b];
    int it = iidx[b];
    float2 ue = ((const float2*)(ui + (size_t)u * D))[lane];
    float2 ie = ((const float2*)(ui + (size_t)(NUM_USERS + it) * D))[lane];

    union { _Float16 h[2]; unsigned int u32; } pk;
    unsigned int* drow = (unsigned int*)(cat16 + (size_t)b * 128);
    pk.h[0] = (_Float16)ue.x; pk.h[1] = (_Float16)ue.y;
    drow[lane] = pk.u32;                 // cols [0,64)
    pk.h[0] = (_Float16)ie.x; pk.h[1] = (_Float16)ie.y;
    drow[32 + lane] = pk.u32;            // cols [64,128)

    float2 w = ((const float2*)aff_w)[lane];
    float p = ue.x * ie.x * w.x + ue.y * ie.y * w.y;
    #pragma unroll
    for (int m = 16; m; m >>= 1) p += __shfl_xor(p, m, 32);
    if (lane == 0) logits[b] = p + aff_b[0];
}

// ---------------------------------------------------------------------------
// 6) Pack mlp_w f32 [128 x 256] into WMMA B-fragment layout (f16 pairs):
//    dword index = ((nt*4 + kcIdx)*32 + lane)*8 + r
//    holds (K = kcIdx*32 + (lane>>4)*16 + 2r, N = nt*16 + (lane&15)) pair.
//    GEMM then loads each lane fragment as 2 contiguous b128.
// ---------------------------------------------------------------------------
__global__ void k_wpack(const float* __restrict__ w, unsigned int* __restrict__ pb) {
    int i = blockIdx.x * blockDim.x + threadIdx.x;
    if (i >= 16 * 4 * 32 * 8) return;
    int r     = i & 7;
    int lane  = (i >> 3) & 31;
    int kcIdx = (i >> 8) & 3;
    int nt    = i >> 10;
    int n  = nt * 16 + (lane & 15);
    int k0 = kcIdx * 32 + (lane >> 4) * 16 + 2 * r;
    union { _Float16 h[2]; unsigned int u32; } pk;
    pk.h[0] = (_Float16)w[(size_t)k0 * 256 + n];
    pk.h[1] = (_Float16)w[(size_t)(k0 + 1) * 256 + n];
    pb[i] = pk.u32;
}

// ---------------------------------------------------------------------------
// 7) WMMA MLP:  h = relu(cat @ mlp_w + mlp_b), out = sigmoid(logits + h@mlp1_w + b)
//    Block = 16 batch rows, 8 waves; wave owns cols [32w, 32w+32) (2 tiles).
//    A tile (16x128 f16 = 4KB) staged to LDS via async global->LDS b128,
//    all waves then read it with ds_load_b128.  B from pre-packed fragments.
// ---------------------------------------------------------------------------
__global__ void __launch_bounds__(256)
k_mlp(const _Float16* __restrict__ cat16,
      const unsigned int* __restrict__ pb,
      const float* __restrict__ mlp_b,
      const float* __restrict__ mlp1_w,
      const float* __restrict__ mlp1_b,
      const float* __restrict__ logits,
      float* __restrict__ out) {
    __shared__ float red[16];
    __shared__ __align__(16) _Float16 tileA[16 * 128];

    int wave = threadIdx.x >> 5;
    int lane = threadIdx.x & 31;
    int row0 = blockIdx.x * 16;

    if (threadIdx.x < 16) red[threadIdx.x] = 0.0f;

    // ---- async copy: 256 threads x 16B = 4KB A tile into LDS ----
    {
        const _Float16* gsrc = cat16 + (size_t)row0 * 128 + threadIdx.x * 8;
        unsigned ldsoff = (unsigned)(size_t)(tileA + threadIdx.x * 8);
        asm volatile("global_load_async_to_lds_b128 %0, %1, off"
                     :: "v"(ldsoff), "v"(gsrc) : "memory");
        asm volatile("s_wait_asynccnt 0" ::: "memory");
    }
    __syncthreads();

    int m  = lane & 15;        // A row / B-C column within tile
    int hi = lane >> 4;        // lane-half selector

    const _Float16* arow = tileA + m * 128;

    v8f c0 = {};
    v8f c1 = {};

    #pragma unroll
    for (int kcIdx = 0; kcIdx < 4; ++kcIdx) {
        int kc = kcIdx * 32;
        // A fragment: two aligned 16B LDS reads per lane
        union { v16h v; uint4 q[2]; } a;
        a.q[0] = *(const uint4*)(arow + kc + hi * 16);
        a.q[1] = *(const uint4*)(arow + kc + hi * 16 + 8);
        #pragma unroll
        for (int t = 0; t < 2; ++t) {
            const unsigned int* bp =
                pb + ((((size_t)(wave * 2 + t) * 4 + kcIdx) * 32 + lane) * 8);
            union { v16h v; uint4 q[2]; } b;
            b.q[0] = *(const uint4*)(bp);
            b.q[1] = *(const uint4*)(bp + 4);
            if (t == 0)
                c0 = __builtin_amdgcn_wmma_f32_16x16x32_f16(
                        false, a.v, false, b.v, (short)0, c0, false, false);
            else
                c1 = __builtin_amdgcn_wmma_f32_16x16x32_f16(
                        false, a.v, false, b.v, (short)0, c1, false, false);
        }
    }

    // ---- Epilogue: relu(+bias), fold GEMV with mlp1_w via LDS f32 atomics ----
    // C/D layout: VGPR r, lanes 0-15 -> (M=r, N=lane); lanes 16-31 -> (M=8+r, N=lane-16)
    int n0 = wave * 32 + m;
    int n1 = wave * 32 + 16 + m;
    float w1n0 = mlp1_w[n0];
    float w1n1 = mlp1_w[n1];
    float b0 = mlp_b[n0];
    float b1 = mlp_b[n1];
    #pragma unroll
    for (int r = 0; r < 8; ++r) {
        int mrow = hi * 8 + r;
        float h0 = fmaxf(c0[r] + b0, 0.0f);
        float h1 = fmaxf(c1[r] + b1, 0.0f);
        atomicAdd(&red[mrow], h0 * w1n0 + h1 * w1n1);   // ds_add_f32
    }
    __syncthreads();

    if (threadIdx.x < 16) {
        float z = logits[row0 + threadIdx.x] + red[threadIdx.x] + mlp1_b[0];
        out[row0 + threadIdx.x] = 1.0f / (1.0f + expf(-z));
    }
}

// ---------------------------------------------------------------------------
extern "C" void kernel_launch(void* const* d_in, const int* in_sizes, int n_in,
                              void* d_out, int out_size, void* d_ws, size_t ws_size,
                              hipStream_t stream) {
    const float* user_w   = (const float*)d_in[0];
    const float* item_w   = (const float*)d_in[1];
    const float* edge_val = (const float*)d_in[2];
    const float* affine_w = (const float*)d_in[3];
    const float* affine_b = (const float*)d_in[4];
    const float* mlp_w    = (const float*)d_in[5];
    const float* mlp_b    = (const float*)d_in[6];
    const float* mlp1_w   = (const float*)d_in[7];
    const float* mlp1_b   = (const float*)d_in[8];
    const int*   edge_row = (const int*)d_in[9];
    const int*   edge_col = (const int*)d_in[10];
    const int*   user_idx = (const int*)d_in[11];
    const int*   item_idx = (const int*)d_in[12];
    float* out = (float*)d_out;

    char* ws = (char*)d_ws;
    size_t nodeBytes = (size_t)N_NODES * D * sizeof(float);
    float*        ego    = (float*)ws;         ws += nodeBytes;
    float*        agg    = (float*)ws;         ws += nodeBytes;
    float*        acc    = (float*)ws;         ws += nodeBytes;
    _Float16*     cat16  = (_Float16*)ws;      ws += (size_t)BATCH * 128 * sizeof(_Float16);
    unsigned int* pb     = (unsigned int*)ws;  ws += (size_t)16 * 4 * 32 * 8 * sizeof(unsigned int);
    float*        logits = (float*)ws;         ws += (size_t)BATCH * sizeof(float);

    k_normalize_init<<<(N_NODES + 7) / 8, 256, 0, stream>>>(user_w, item_w, ego, acc);

    for (int l = 0; l < N_LAYERS; ++l) {
        hipMemsetAsync(agg, 0, nodeBytes, stream);
        k_spmm<<<(N_EDGES + 7) / 8, 256, 0, stream>>>(ego, edge_row, edge_col, edge_val, agg);
        k_combine<<<(N_NODES * D + 255) / 256, 256, 0, stream>>>(agg, ego, acc, N_NODES * D);
    }

    k_final_norm<<<(N_NODES + 7) / 8, 256, 0, stream>>>(acc, ego);  // ego now holds ui
    k_prep<<<BATCH / 8, 256, 0, stream>>>(ego, user_idx, item_idx, affine_w, affine_b,
                                          cat16, logits);
    k_wpack<<<(16 * 4 * 32 * 8 + 255) / 256, 256, 0, stream>>>(mlp_w, pb);
    k_mlp<<<BATCH / 16, 256, 0, stream>>>(cat16, pb, mlp_b, mlp1_w, mlp1_b, logits, out);
}
